// DMMModel_6055903887493
// MI455X (gfx1250) — matrix operand order repaired and checked by reading, not compiled, not run on hardware
//
#include <hip/hip_runtime.h>
#include <hip/hip_bf16.h>
#include <cstdint>

// ---------------------------------------------------------------------------
// DMM model, fused for MI455X (gfx1250, wave32, WMMA).
//   B=512, T=128, H=256, E=F=128, D=E+F=256, V=64
// LSTM(reversed) and DMM scans run in the same t' order and are per-batch-row
// independent -> one fused loop, 16 rows per workgroup, 32 WGs.
// GEMMs use v_wmma_f32_16x16x32_f16 with weights pre-packed into a contiguous-k WMMA
// B-fragment layout (lane%16 = N, (lane/16)*16+e = K). The K-panel loop is
// software-pipelined one panel deep with TPW distinct B-register buffers
// (slot index is always a compile-time constant -> no spills), so weight
// loads for panel kt+1 overlap the WMMAs of panel kt.
// ---------------------------------------------------------------------------

typedef __attribute__((ext_vector_type(16))) _Float16 v16h;
typedef __attribute__((ext_vector_type(8)))  _Float16 v8h;
typedef __attribute__((ext_vector_type(8)))  float    v8f;

union AF { v16h v; v8h h[2]; };

constexpr int Bb = 512;   // batch
constexpr int Tt = 128;   // time
constexpr int Hh = 256;   // hidden
constexpr int Ee = 128;   // embed dim
constexpr int Ff = 128;   // feature len
constexpr int Dd = 256;   // E+F

// Packed weight element counts (f16)
constexpr int W_LSTM_ELEMS = 512 * 1024;  // [x;h](K=512) -> 4H(N=1024)
constexpr int W_INF_ELEMS  = 512 * 512;   // [g;z](K=512) -> [mu|lv](N=512)
constexpr int W_DEC_ELEMS  = 256 * 512;   // z(K=256)     -> [mu|lv](N=512)

__device__ __forceinline__ float sigmoid_fast(float x) {
  return 1.0f / (1.0f + __expf(-x));
}
__device__ __forceinline__ float tanh_fast(float x) {
  float e = __expf(2.0f * x);
  return (e - 1.0f) / (e + 1.0f);
}

// ---------------------------------------------------------------------------
// Weight repack: fp32 row-major -> f16 WMMA B-tile layout.
// Tile (kt,nt): 32 lanes x 16 contiguous f16.  Element (L,e) holds
//   W[kt*32 + (L>>4)*16 + e][nt*16 + (L&15)]
// ---------------------------------------------------------------------------
__global__ __launch_bounds__(256) void repack_weights(
    const float* __restrict__ lstm_k,  // [256,1024]
    const float* __restrict__ lstm_rk, // [256,1024]
    const float* __restrict__ inf_Wm,  // [512,256]
    const float* __restrict__ inf_Wv,  // [512,256]
    const float* __restrict__ dec_Wm,  // [256,256]
    const float* __restrict__ dec_Wv,  // [256,256]
    _Float16* __restrict__ Wl,
    _Float16* __restrict__ Wi,
    _Float16* __restrict__ Wd)
{
  int idx = blockIdx.x * blockDim.x + threadIdx.x;
  if (idx < W_LSTM_ELEMS) {
    int tile = idx >> 9, rem = idx & 511;           // 512 elems per tile
    int L = rem >> 4, e = rem & 15;
    int kt = tile >> 6, nt = tile & 63;             // NT = 64
    int K = kt * 32 + ((L >> 4) << 4) + e;
    int N = nt * 16 + (L & 15);
    float v = (K < 256) ? lstm_k[K * 1024 + N] : lstm_rk[(K - 256) * 1024 + N];
    Wl[idx] = (_Float16)v;
    return;
  }
  idx -= W_LSTM_ELEMS;
  if (idx < W_INF_ELEMS) {
    int tile = idx >> 9, rem = idx & 511;
    int L = rem >> 4, e = rem & 15;
    int kt = tile >> 5, nt = tile & 31;             // NT = 32
    int K = kt * 32 + ((L >> 4) << 4) + e;
    int N = nt * 16 + (L & 15);
    float v = (N < 256) ? inf_Wm[K * 256 + N] : inf_Wv[K * 256 + (N - 256)];
    Wi[idx] = (_Float16)v;
    return;
  }
  idx -= W_INF_ELEMS;
  if (idx < W_DEC_ELEMS) {
    int tile = idx >> 9, rem = idx & 511;
    int L = rem >> 4, e = rem & 15;
    int kt = tile >> 5, nt = tile & 31;             // NT = 32
    int K = kt * 32 + ((L >> 4) << 4) + e;
    int N = nt * 16 + (L & 15);
    float v = (N < 256) ? dec_Wm[K * 256 + N] : dec_Wv[K * 256 + (N - 256)];
    Wd[idx] = (_Float16)v;
  }
}

// ---------------------------------------------------------------------------
// One GEMM phase: D[16 x NT*16] = A[16 x KT*32] @ Wpacked, f32 accum in VGPRs,
// result scattered to LDS sbuf (row stride OSTR). A rows live in LDS with row
// stride 512 f16; acol offsets into the K window.
//
// Pipelining: dynamic kt loop, manually unrolled 2x (a0/a1 A-fragments),
// TPW B-fragment buffers refilled one panel ahead right after consumption.
// All register-array indices are compile-time constants.
// ---------------------------------------------------------------------------
template<int KT, int TPW, int OSTR>
__device__ __forceinline__ void gemm_phase(
    const _Float16* __restrict__ A, int acol,
    const _Float16* __restrict__ W, int NT,
    float* __restrict__ sb, int lane, int wave)
{
  static_assert((KT & 1) == 0, "KT must be even");

  const v8f vzero = {0.f, 0.f, 0.f, 0.f, 0.f, 0.f, 0.f, 0.f};
  v8f acc[TPW];
#pragma unroll
  for (int j = 0; j < TPW; ++j) acc[j] = vzero;

  const int M  = lane & 15;
  const int kh = lane >> 4;
  const _Float16* Abase = A + M * 512 + acol + kh * 16;
  const _Float16* Wbase = W + ((size_t)(wave * TPW) * 32 + lane) * 16;

  AF a0, a1;
  AF b[TPW];

  auto ldA = [&](AF& af, int kt) {
    const _Float16* ap = Abase + kt * 32;
    af.h[0] = *(const v8h*)(ap);
    af.h[1] = *(const v8h*)(ap + 8);
  };
  auto ldB = [&](AF& bf, int kt, int j) {
    const _Float16* bp = Wbase + ((size_t)kt * NT + j) * 512;
    bf.h[0] = *(const v8h*)(bp);
    bf.h[1] = *(const v8h*)(bp + 8);
  };

  // prologue: A panel 0 + B panel 0 in flight
  ldA(a0, 0);
#pragma unroll
  for (int j = 0; j < TPW; ++j) ldB(b[j], 0, j);

#pragma clang loop unroll(disable)
  for (int kt = 0; kt < KT; kt += 2) {
    // half 1: consume panel kt, refill with panel kt+1
    ldA(a1, kt + 1);
#pragma unroll
    for (int j = 0; j < TPW; ++j) {
      acc[j] = __builtin_amdgcn_wmma_f32_16x16x32_f16(
          false, a0.v, false, b[j].v, (short)0, acc[j], false, false);
      ldB(b[j], kt + 1, j);
    }
    // half 2: consume panel kt+1, refill with panel kt+2 (clamped, branch-free)
    const int ktn = (kt + 2 < KT) ? (kt + 2) : (KT - 2);
    ldA(a0, ktn);
#pragma unroll
    for (int j = 0; j < TPW; ++j) {
      acc[j] = __builtin_amdgcn_wmma_f32_16x16x32_f16(
          false, a1.v, false, b[j].v, (short)0, acc[j], false, false);
      ldB(b[j], ktn, j);
    }
  }

#pragma unroll
  for (int j = 0; j < TPW; ++j) {
    int col = (wave * TPW + j) * 16 + M;
#pragma unroll
    for (int r = 0; r < 8; ++r) {
      int row = kh * 8 + r;
      sb[row * OSTR + col] = acc[j][r];
    }
  }
}

// ---------------------------------------------------------------------------
// Fused model kernel: grid = B/16 workgroups, 256 threads (8 waves).
// Dynamic LDS: aLDS [16][512] f16 (x | h), cbuf [16][512] f16 (h | z),
//              sbuf [16][1024] f32 (GEMM staging / gate preacts).
// ---------------------------------------------------------------------------
__global__ __launch_bounds__(256) void dmm_fused(
    const int*   __restrict__ events,   // [B,T]
    const float* __restrict__ features, // [B,T,F]
    const float* __restrict__ emb,      // [V,E]
    const float* __restrict__ lstm_b,   // [4H]
    const float* __restrict__ inf_bm,   // [H]
    const float* __restrict__ inf_bv,   // [H]
    const float* __restrict__ dec_bm,   // [D]
    const float* __restrict__ dec_bv,   // [D]
    const float* __restrict__ z0,       // [B,H]
    const float* __restrict__ eps_inf,  // [T,B,H]
    const float* __restrict__ eps_rec,  // [T,B,D]
    const _Float16* __restrict__ Wl,    // packed LSTM [512->1024]
    const _Float16* __restrict__ Wi,    // packed inf  [512->512]
    const _Float16* __restrict__ Wd,    // packed dec  [256->512]
    float* __restrict__ out)            // [B,D]
{
  extern __shared__ char smem[];
  _Float16* aLDS = (_Float16*)smem;                   // 16*512 f16 = 16 KB
  _Float16* cbuf = aLDS + 16 * 512;                   // 16*512 f16 = 16 KB
  float*    sbuf = (float*)(smem + 2 * 16 * 512 * 2); // 16*1024 f32 = 64 KB

  const int tid  = threadIdx.x;
  const int lane = tid & 31;
  const int wave = tid >> 5;
  const int b0   = blockIdx.x * 16;

  // c-state in registers: thread tid owns fixed (row,col) pairs.
  float creg[16];
#pragma unroll
  for (int e = 0; e < 16; ++e) creg[e] = 0.0f;

  // init: h = 0 (A-buffer high half), z = z0 (comb high half)
#pragma unroll
  for (int e = 0; e < 16; ++e) {
    int idx = tid * 16 + e;
    int row = idx >> 8, col = idx & 255;
    aLDS[row * 512 + 256 + col] = (_Float16)0.0f;
    cbuf[row * 512 + 256 + col] = (_Float16)z0[(b0 + row) * Hh + col];
  }
  __syncthreads();

  for (int t = 0; t < Tt; ++t) {
    const int torig = Tt - 1 - t;  // go_backwards

    // phase 0: build x_t = [emb[events] | features] as f16 into aLDS[:, :256]
#pragma unroll
    for (int e = 0; e < 16; ++e) {
      int idx = tid * 16 + e;
      int row = idx >> 8, col = idx & 255;
      int b = b0 + row;
      float v;
      if (col < Ee) {
        int ev = events[b * Tt + torig];
        v = emb[ev * Ee + col];
      } else {
        v = features[((size_t)b * Tt + torig) * Ff + (col - Ee)];
      }
      aLDS[row * 512 + col] = (_Float16)v;
    }
    __syncthreads();

    // phase 1: zg = [x|h] @ [Wk;Wrk]  (16 x 512 x 1024)
    gemm_phase<16, 8, 1024>(aLDS, 0, Wl, 64, sbuf, lane, wave);
    __syncthreads();

    // phase 2: gates (i,f,g,o), c/h update; h -> aLDS high half & cbuf low half
#pragma unroll
    for (int e = 0; e < 16; ++e) {
      int idx = tid * 16 + e;
      int row = idx >> 8, col = idx & 255;
      float zi = sbuf[row * 1024 + col]       + lstm_b[col];
      float zf = sbuf[row * 1024 + 256 + col] + lstm_b[256 + col];
      float zg = sbuf[row * 1024 + 512 + col] + lstm_b[512 + col];
      float zo = sbuf[row * 1024 + 768 + col] + lstm_b[768 + col];
      float ig = sigmoid_fast(zi);
      float fg = sigmoid_fast(zf);
      float gg = tanh_fast(zg);
      float og = sigmoid_fast(zo);
      float c = fg * creg[e] + ig * gg;
      creg[e] = c;
      float h = og * tanh_fast(c);
      aLDS[row * 512 + 256 + col] = (_Float16)h;
      cbuf[row * 512 + col]       = (_Float16)h;
    }
    __syncthreads();

    // phase 3: [mu|lv] = [g|z_prev] @ [Wm|Wv]  (16 x 512 x 512)
    gemm_phase<16, 4, 512>(cbuf, 0, Wi, 32, sbuf, lane, wave);
    __syncthreads();

    // phase 4: z = mu + exp(0.5*lv) * eps_inf[t]; store f16 into cbuf high half
#pragma unroll
    for (int e = 0; e < 16; ++e) {
      int idx = tid * 16 + e;
      int row = idx >> 8, col = idx & 255;
      int b = b0 + row;
      float mu = sbuf[row * 512 + col]       + inf_bm[col];
      float lv = sbuf[row * 512 + 256 + col] + inf_bv[col];
      float z = mu + __expf(0.5f * lv) * eps_inf[((size_t)t * Bb + b) * Hh + col];
      cbuf[row * 512 + 256 + col] = (_Float16)z;
    }
    __syncthreads();
  }

  // decoder (last step only): [x_mu|x_lv] = z @ [dec_Wm|dec_Wv] (16 x 256 x 512)
  gemm_phase<8, 4, 512>(cbuf, 256, Wd, 32, sbuf, lane, wave);
  __syncthreads();

#pragma unroll
  for (int e = 0; e < 16; ++e) {
    int idx = tid * 16 + e;
    int row = idx >> 8, col = idx & 255;
    int b = b0 + row;
    float xm = sbuf[row * 512 + col]       + dec_bm[col];
    float xl = sbuf[row * 512 + 256 + col] + dec_bv[col];
    out[(size_t)b * Dd + col] =
        xm + __expf(0.5f * xl) * eps_rec[((size_t)(Tt - 1) * Bb + b) * Dd + col];
  }
}

// ---------------------------------------------------------------------------
extern "C" void kernel_launch(void* const* d_in, const int* in_sizes, int n_in,
                              void* d_out, int out_size, void* d_ws, size_t ws_size,
                              hipStream_t stream) {
  (void)in_sizes; (void)n_in; (void)out_size; (void)ws_size;

  const int*   events   = (const int*)  d_in[0];
  const float* features = (const float*)d_in[1];
  const float* emb      = (const float*)d_in[2];
  const float* lstm_k   = (const float*)d_in[3];
  const float* lstm_rk  = (const float*)d_in[4];
  const float* lstm_b   = (const float*)d_in[5];
  const float* inf_Wm   = (const float*)d_in[6];
  const float* inf_bm   = (const float*)d_in[7];
  const float* inf_Wv   = (const float*)d_in[8];
  const float* inf_bv   = (const float*)d_in[9];
  const float* dec_Wm   = (const float*)d_in[10];
  const float* dec_bm   = (const float*)d_in[11];
  const float* dec_Wv   = (const float*)d_in[12];
  const float* dec_bv   = (const float*)d_in[13];
  const float* z0       = (const float*)d_in[14];
  const float* eps_inf  = (const float*)d_in[15];
  const float* eps_rec  = (const float*)d_in[16];
  float* out = (float*)d_out;

  _Float16* Wl = (_Float16*)d_ws;
  _Float16* Wi = Wl + W_LSTM_ELEMS;
  _Float16* Wd = Wi + W_INF_ELEMS;

  // 1) repack weights into WMMA B-fragment layout (917504 elems)
  {
    int total = W_LSTM_ELEMS + W_INF_ELEMS + W_DEC_ELEMS;
    int blocks = (total + 255) / 256;
    repack_weights<<<blocks, 256, 0, stream>>>(
        lstm_k, lstm_rk, inf_Wm, inf_Wv, dec_Wm, dec_Wv, Wl, Wi, Wd);
  }

  // 2) fused recurrent model: 32 WGs x 256 threads, 96 KB dynamic LDS
  {
    size_t smem = (size_t)2 * 16 * 512 * 2 /*a,c f16*/ + (size_t)16 * 1024 * 4 /*sbuf*/;
    dmm_fused<<<Bb / 16, 256, smem, stream>>>(
        events, features, emb, lstm_b, inf_bm, inf_bv, dec_bm, dec_bv,
        z0, eps_inf, eps_rec, Wl, Wi, Wd, out);
  }
}
